// PointAttentionBlock_78219944394880
// MI455X (gfx1250) — matrix operand order repaired and checked by reading, not compiled, not run on hardware
//
#include <hip/hip_runtime.h>
#include <hip/hip_bf16.h>

typedef __attribute__((ext_vector_type(16))) _Float16 v16h;
typedef __attribute__((ext_vector_type(8)))  _Float16 v8h;
typedef __attribute__((ext_vector_type(8)))  float    v8f;

#define NB   4
#define CC   256
#define NN   4096
#define CQ   32

// ---------------------------------------------------------------------------
// WMMA operand loaders using this code's contiguous k order for both A and B (not the ISA table order)
// A (16x32 f16, M x K): lane m=L%16, g=L/16; elem e -> K=g*16+e
// => one contiguous 32B run from row-major storage (rows = M, contiguous K).
// ---------------------------------------------------------------------------
__device__ __forceinline__ v16h load_A16(const _Float16* __restrict__ base,
                                         int stride_h, int m_base, int lane) {
  const int m = m_base + (lane & 15);
  const int g = (lane >> 4) & 1;
  const _Float16* p = base + (size_t)m * stride_h + g * 16;
  v8h lo = *(const v8h*)(p);        // K = g*16 + 0..7
  v8h hi = *(const v8h*)(p + 8);    // K = g*16 + 8..15
  v16h a;
#pragma unroll
  for (int e = 0; e < 8; ++e) { a[e] = lo[e]; a[e + 8] = hi[e]; }
  return a;
}

// B (32x16 f16, K x N) where B[k][n] = rowmajor[n][k]:
// lane n=L%16, g=L/16; elem e -> K = g*16 + e => one contiguous 32B run per lane.
__device__ __forceinline__ v16h load_B16(const _Float16* __restrict__ base,
                                         int stride_h, int n_base, int lane) {
  const int n = n_base + (lane & 15);
  const int g = (lane >> 4) & 1;
  const _Float16* p = base + (size_t)n * stride_h + g * 16;
  v8h lo = *(const v8h*)(p);
  v8h hi = *(const v8h*)(p + 8);
  v16h b;
#pragma unroll
  for (int e = 0; e < 8; ++e) { b[e] = lo[e]; b[e + 8] = hi[e]; }
  return b;
}

__device__ __forceinline__ v8f wmma_f16(v16h a, v16h b, v8f c) {
  return __builtin_amdgcn_wmma_f32_16x16x32_f16(false, a, false, b,
                                                (short)0, c, false, false);
}

// ---------------------------------------------------------------------------
// f32 -> f16 elementwise cast (weights)
// ---------------------------------------------------------------------------
__global__ void cvt_f16_kernel(const float* __restrict__ s,
                               _Float16* __restrict__ d, int n) {
  int i = blockIdx.x * blockDim.x + threadIdx.x;
  if (i < n) d[i] = (_Float16)s[i];
}

// ---------------------------------------------------------------------------
// x [B,C,N] f32 -> xT [B,N,C] f16, 32x32 LDS tile transpose
// ---------------------------------------------------------------------------
__global__ void xpose_f16_kernel(const float* __restrict__ x,
                                 _Float16* __restrict__ xt) {
  __shared__ float t[32][33];
  const int b = blockIdx.z, c0 = blockIdx.y * 32, n0 = blockIdx.x * 32;
  const int col = threadIdx.x & 31, rb = threadIdx.x >> 5;
  const float* xb = x + (size_t)b * CC * NN;
#pragma unroll
  for (int it = 0; it < 4; ++it) {
    int r = rb + it * 8;
    t[r][col] = xb[(size_t)(c0 + r) * NN + (n0 + col)];
  }
  __syncthreads();
  _Float16* xtb = xt + (size_t)b * NN * CC;
#pragma unroll
  for (int it = 0; it < 4; ++it) {
    int r = rb + it * 8;
    xtb[(size_t)(n0 + r) * CC + (c0 + col)] = (_Float16)t[col][r];
  }
}

// ---------------------------------------------------------------------------
// Q/K projection: Q[b,n,o] = sum_c xT[b,n,c]*wq[o,c] + bq[o]   (o < 32)
// One wave per 16-row n-tile; A from xT, B from weight rows; 32 WMMAs/wave.
// ---------------------------------------------------------------------------
__global__ void __launch_bounds__(256) qk_proj_kernel(
    const _Float16* __restrict__ xt, const _Float16* __restrict__ wqh,
    const _Float16* __restrict__ wkh, const float* __restrict__ bq,
    const float* __restrict__ bk, _Float16* __restrict__ qh,
    _Float16* __restrict__ kh) {
  const int lane = threadIdx.x & 31;
  const int wave = blockIdx.x * (blockDim.x >> 5) + (threadIdx.x >> 5);
  const int b = wave >> 8;          // 256 n-tiles per batch
  const int n_base = (wave & 255) * 16;
  const _Float16* xtb = xt + (size_t)b * NN * CC;
  v8f aq0 = {}, aq1 = {}, ak0 = {}, ak1 = {};
#pragma unroll
  for (int kc = 0; kc < 8; ++kc) {
    v16h A   = load_A16(xtb + kc * 32, CC, n_base, lane);
    v16h Bq0 = load_B16(wqh + kc * 32, CC, 0,  lane);
    v16h Bq1 = load_B16(wqh + kc * 32, CC, 16, lane);
    v16h Bk0 = load_B16(wkh + kc * 32, CC, 0,  lane);
    v16h Bk1 = load_B16(wkh + kc * 32, CC, 16, lane);
    aq0 = wmma_f16(A, Bq0, aq0);  aq1 = wmma_f16(A, Bq1, aq1);
    ak0 = wmma_f16(A, Bk0, ak0);  ak1 = wmma_f16(A, Bk1, ak1);
  }
  const int g = lane >> 4, li = lane & 15;
  _Float16* qb = qh + (size_t)b * NN * CQ;
  _Float16* kb = kh + (size_t)b * NN * CQ;
  const float b0q = bq[li], b1q = bq[16 + li];
  const float b0k = bk[li], b1k = bk[16 + li];
#pragma unroll
  for (int r = 0; r < 8; ++r) {
    int n = n_base + r + 8 * g;
    qb[(size_t)n * CQ + li]      = (_Float16)(aq0[r] + b0q);
    qb[(size_t)n * CQ + 16 + li] = (_Float16)(aq1[r] + b1q);
    kb[(size_t)n * CQ + li]      = (_Float16)(ak0[r] + b0k);
    kb[(size_t)n * CQ + 16 + li] = (_Float16)(ak1[r] + b1k);
  }
}

// ---------------------------------------------------------------------------
// V projection: V[b,co,n] = sum_c wv[co,c]*xT[b,n,c] + bv[co]  stored [B,C,N] f16
// One wave per 16x16 output tile; A from wv rows, B from xT rows.
// ---------------------------------------------------------------------------
__global__ void __launch_bounds__(256) v_proj_kernel(
    const _Float16* __restrict__ xt, const _Float16* __restrict__ wvh,
    const float* __restrict__ bv, _Float16* __restrict__ vh) {
  const int lane = threadIdx.x & 31;
  const int wave = blockIdx.x * (blockDim.x >> 5) + (threadIdx.x >> 5);
  const int n_base = (wave & 255) * 16;
  const int c_base = ((wave >> 8) & 15) * 16;
  const int b = wave >> 12;
  const _Float16* xtb = xt + (size_t)b * NN * CC;
  v8f acc = {};
#pragma unroll
  for (int kc = 0; kc < 8; ++kc) {
    v16h A = load_A16(wvh + kc * 32, CC, c_base, lane);
    v16h B = load_B16(xtb + kc * 32, CC, n_base, lane);
    acc = wmma_f16(A, B, acc);
  }
  const int g = lane >> 4, li = lane & 15;
  _Float16* vb = vh + (size_t)b * CC * NN;
  const int n = n_base + li;
#pragma unroll
  for (int r = 0; r < 8; ++r) {
    int c = c_base + r + 8 * g;
    vb[(size_t)c * NN + n] = (_Float16)(acc[r] + bv[c]);
  }
}

// ---------------------------------------------------------------------------
// Flash attention, S^T orientation, 32 queries per wave (two query tiles).
// Each lane owns one query column per tile -> softmax stats are per-lane
// scalars; P^T -> B-operand repack needs only a cross-half shfl_xor(16).
// Each V A-tile (2 x b128 loads) feeds TWO independent WMMAs (one per query
// tile), halving V/K traffic per query vs a 16-query tile.
// Accumulator rescale is gated on a wave-uniform ballot (scalar branch, EXEC
// stays all-ones as WMMA requires). Epilogue fuses out = alpha*O/l + x.
// ---------------------------------------------------------------------------
__global__ void __launch_bounds__(256, 1) attn_kernel(
    const _Float16* __restrict__ qh, const _Float16* __restrict__ kh,
    const _Float16* __restrict__ vh, const float* __restrict__ x,
    const float* __restrict__ alpha_p, float* __restrict__ out) {
  const int lane = threadIdx.x & 31;
  const int wave = blockIdx.x * (blockDim.x >> 5) + (threadIdx.x >> 5);
  const int b = wave >> 7;              // 128 query-pair tiles per batch
  const int i_base = (wave & 127) * 32; // 32 queries per wave
  const int g = lane >> 4, li = lane & 15;

  const _Float16* qb = qh + (size_t)b * NN * CQ;
  const _Float16* kb = kh + (size_t)b * NN * CQ;
  const _Float16* vb = vh + (size_t)b * CC * NN;

  // Loop-invariant Q as B operands: B[k=d][n=i] = Q[i][d]
  const v16h Qb0 = load_B16(qb, CQ, i_base, lane);
  const v16h Qb1 = load_B16(qb, CQ, i_base + 16, lane);

  v8f acc0[16], acc1[16];
#pragma unroll
  for (int t = 0; t < 16; ++t) { acc0[t] = (v8f){}; acc1[t] = (v8f){}; }
  float m0 = -3.0e38f, l0 = 0.0f;
  float m1 = -3.0e38f, l1 = 0.0f;
  const v8f zero = {};

  for (int j0 = 0; j0 < NN; j0 += 32) {
    if (j0 + 32 < NN)  // prefetch next K tile -> global_prefetch_b8
      __builtin_prefetch(kb + (size_t)(j0 + 32) * CQ, 0, 1);

    const v16h K0 = load_A16(kb, CQ, j0, lane);
    const v16h K1 = load_A16(kb, CQ, j0 + 16, lane);

    // ---- query tile 0: S^T, online softmax, pack P^T ----
    v8f T0 = wmma_f16(K0, Qb0, zero);
    v8f T1 = wmma_f16(K1, Qb0, zero);
    float cm = -3.0e38f;
#pragma unroll
    for (int r = 0; r < 8; ++r) { cm = fmaxf(cm, T0[r]); cm = fmaxf(cm, T1[r]); }
    cm = fmaxf(cm, __shfl_xor(cm, 16, 32));
    const float mn0 = fmaxf(m0, cm);
    const float sc0 = __expf(m0 - mn0);
    float P0[8], P1[8], ps = 0.0f;
#pragma unroll
    for (int r = 0; r < 8; ++r) {
      P0[r] = __expf(T0[r] - mn0);
      P1[r] = __expf(T1[r] - mn0);
      ps += P0[r] + P1[r];
    }
    ps += __shfl_xor(ps, 16, 32);
    const bool grew0 = mn0 > m0;
    l0 = l0 * sc0 + ps;
    m0 = mn0;
    v16h Pb0;
#pragma unroll
    for (int e = 0; e < 8; ++e) {
      float t1x = __shfl_xor(P1[e], 16, 32);
      float t0x = __shfl_xor(P0[e], 16, 32);
      Pb0[e]     = (_Float16)(g ? t1x : P0[e]);
      Pb0[e + 8] = (_Float16)(g ? P1[e] : t0x);
    }

    // ---- query tile 1 ----
    T0 = wmma_f16(K0, Qb1, zero);
    T1 = wmma_f16(K1, Qb1, zero);
    cm = -3.0e38f;
#pragma unroll
    for (int r = 0; r < 8; ++r) { cm = fmaxf(cm, T0[r]); cm = fmaxf(cm, T1[r]); }
    cm = fmaxf(cm, __shfl_xor(cm, 16, 32));
    const float mn1 = fmaxf(m1, cm);
    const float sc1 = __expf(m1 - mn1);
    ps = 0.0f;
#pragma unroll
    for (int r = 0; r < 8; ++r) {
      P0[r] = __expf(T0[r] - mn1);
      P1[r] = __expf(T1[r] - mn1);
      ps += P0[r] + P1[r];
    }
    ps += __shfl_xor(ps, 16, 32);
    const bool grew1 = mn1 > m1;
    l1 = l1 * sc1 + ps;
    m1 = mn1;
    v16h Pb1;
#pragma unroll
    for (int e = 0; e < 8; ++e) {
      float t1x = __shfl_xor(P1[e], 16, 32);
      float t0x = __shfl_xor(P0[e], 16, 32);
      Pb1[e]     = (_Float16)(g ? t1x : P0[e]);
      Pb1[e + 8] = (_Float16)(g ? P1[e] : t0x);
    }

    // Rescale only when some lane's max moved (wave-uniform scalar branch;
    // no WMMA inside, EXEC untouched). Lanes whose max is unchanged have
    // scale == 1.0 exactly.
    if (__ballot(grew0)) {
#pragma unroll
      for (int t = 0; t < 16; ++t)
#pragma unroll
        for (int r = 0; r < 8; ++r) acc0[t][r] *= sc0;
    }
    if (__ballot(grew1)) {
#pragma unroll
      for (int t = 0; t < 16; ++t)
#pragma unroll
        for (int r = 0; r < 8; ++r) acc1[t][r] *= sc1;
    }

    // O += V * P^T : each V A-tile feeds both query tiles.
#pragma unroll
    for (int t = 0; t < 16; ++t) {
      v16h Va = load_A16(vb + j0, NN, t * 16, lane);
      acc0[t] = wmma_f16(Va, Pb0, acc0[t]);
      acc1[t] = wmma_f16(Va, Pb1, acc1[t]);
    }
  }

  const float alpha = alpha_p[0];
  const float s0 = alpha / l0;
  const float s1 = alpha / l1;
  const float* xb = x + (size_t)b * CC * NN;
  float* ob = out + (size_t)b * CC * NN;
  const int i0 = i_base + li;
  const int i1 = i_base + 16 + li;
#pragma unroll
  for (int t = 0; t < 16; ++t)
#pragma unroll
    for (int r = 0; r < 8; ++r) {
      const int c = t * 16 + r + 8 * g;
      const size_t o0 = (size_t)c * NN + i0;
      const size_t o1 = (size_t)c * NN + i1;
      ob[o0] = acc0[t][r] * s0 + xb[o0];
      ob[o1] = acc1[t][r] * s1 + xb[o1];
    }
}

// ---------------------------------------------------------------------------
extern "C" void kernel_launch(void* const* d_in, const int* in_sizes, int n_in,
                              void* d_out, int out_size, void* d_ws, size_t ws_size,
                              hipStream_t stream) {
  const float* x     = (const float*)d_in[0];
  const float* wq    = (const float*)d_in[1];
  const float* bq    = (const float*)d_in[2];
  const float* wk    = (const float*)d_in[3];
  const float* bk    = (const float*)d_in[4];
  const float* wv    = (const float*)d_in[5];
  const float* bv    = (const float*)d_in[6];
  const float* alpha = (const float*)d_in[7];
  float* out = (float*)d_out;

  // Workspace carve-up (256B aligned), ~19 MB total.
  char* ws = (char*)d_ws;
  size_t off = 0;
  auto carve = [&](size_t bytes) -> char* {
    char* p = ws + off;
    off += (bytes + 255) & ~(size_t)255;
    return p;
  };
  _Float16* xt  = (_Float16*)carve((size_t)NB * NN * CC * 2);  // x^T  f16
  _Float16* qh  = (_Float16*)carve((size_t)NB * NN * CQ * 2);  // Q    f16
  _Float16* kh  = (_Float16*)carve((size_t)NB * NN * CQ * 2);  // K^T  f16
  _Float16* vh  = (_Float16*)carve((size_t)NB * CC * NN * 2);  // V    f16
  _Float16* wqh = (_Float16*)carve((size_t)CQ * CC * 2);
  _Float16* wkh = (_Float16*)carve((size_t)CQ * CC * 2);
  _Float16* wvh = (_Float16*)carve((size_t)CC * CC * 2);

  // 1) weights -> f16
  cvt_f16_kernel<<<(CQ * CC + 255) / 256, 256, 0, stream>>>(wq, wqh, CQ * CC);
  cvt_f16_kernel<<<(CQ * CC + 255) / 256, 256, 0, stream>>>(wk, wkh, CQ * CC);
  cvt_f16_kernel<<<(CC * CC + 255) / 256, 256, 0, stream>>>(wv, wvh, CC * CC);

  // 2) x [B,C,N] f32 -> xT [B,N,C] f16
  xpose_f16_kernel<<<dim3(NN / 32, CC / 32, NB), 256, 0, stream>>>(x, xt);

  // 3) projections (WMMA)
  qk_proj_kernel<<<(NB * (NN / 16)) / 8, 256, 0, stream>>>(xt, wqh, wkh, bq, bk, qh, kh);
  v_proj_kernel<<<(NB * (CC / 16) * (NN / 16)) / 8, 256, 0, stream>>>(xt, wvh, bv, vh);

  // 4) flash attention + fused residual epilogue (WMMA), 32 queries/wave
  attn_kernel<<<(NB * (NN / 32)) / 8, 256, 0, stream>>>(qh, kh, vh, x, alpha, out);
}
